// GraphSAGE_22574348108036
// MI455X (gfx1250) — matrix operand order repaired and checked by reading, not compiled, not run on hardware
//
#include <hip/hip_runtime.h>
#include <hip/hip_bf16.h>
#include <stdint.h>

// ---------------- problem constants (from reference) ----------------
#define NN 50000    // nodes
#define NE 800000   // edges
#define DIM 128     // feature dim
#define NL 3        // layers
#define NG 512      // graphs
#define NT 10       // target dim
#define HC (NL * DIM)   // 384, hcat width
#define LDSB 136        // padded LDS row stride in bf16 elems: 272B = 68 dwords -> bank stride 4

typedef __attribute__((ext_vector_type(16))) __bf16         v16bf;
typedef __attribute__((ext_vector_type(8)))  float          v8f;
typedef __attribute__((ext_vector_type(4)))  unsigned int   v4ui;
typedef __attribute__((ext_vector_type(8)))  unsigned int   v8ui;

#if defined(__has_builtin)
#if __has_builtin(__builtin_amdgcn_cvt_pk_bf16_f32)
#define HAVE_PK_BF16 1
#endif
#endif

// convert 2 f32 -> packed bf16 dword (RNE)
__device__ __forceinline__ unsigned int cvt2_bf16(float a, float b) {
#ifdef HAVE_PK_BF16
  typedef __attribute__((ext_vector_type(2))) __bf16 v2bf;
  v2bf p = __builtin_amdgcn_cvt_pk_bf16_f32(a, b);
  return __builtin_bit_cast(unsigned int, p);
#else
  unsigned ua = __float_as_uint(a);
  unsigned ub = __float_as_uint(b);
  ua += 0x7FFFu + ((ua >> 16) & 1u);
  ub += 0x7FFFu + ((ub >> 16) & 1u);
  // take bytes[3:2] of ua as low half, bytes[3:2] of ub as high half
  return __builtin_amdgcn_perm(ub, ua, 0x07060302u);
#endif
}

__device__ __forceinline__ unsigned short f32_bf16(float f) {
  unsigned u = __float_as_uint(f);
  u += 0x7FFFu + ((u >> 16) & 1u);
  return (unsigned short)(u >> 16);
}

// convert 8 consecutive f32 (scaled) -> 4 dwords of packed bf16
__device__ __forceinline__ v4ui cvt8_bf16(const float* __restrict__ p, float s) {
  v4ui r;
#pragma unroll
  for (int j = 0; j < 4; ++j) r[j] = cvt2_bf16(p[2 * j] * s, p[2 * j + 1] * s);
  return r;
}

// ---------------- weight pre-pack: f32 [K,128] -> bf16 B-fragment layout ----------------
// 16-bit B 32x16 (ISA 7.12.2): lane n(0..15) holds col n, K=kbase+{0..7,16..23};
// lanes 16..31 hold K=kbase+{8..15,24..31}.  flat: frag = ntile*(K/32)+kk; elem = frag*512+lane*16+j
__global__ void pack_b_kernel(const float* __restrict__ W, int K,
                              unsigned short* __restrict__ out) {
  int t = blockIdx.x * 256 + threadIdx.x;
  if (t >= K * DIM) return;
  int KK     = K >> 5;
  int fid    = t >> 9;
  int within = t & 511;
  int lane   = within >> 4;
  int j      = within & 15;
  int kk     = fid % KK;
  int ntile  = fid / KK;
  int k = kk * 32 + ((lane >> 4) << 3) + (j & 7) + ((j >> 3) << 4);
  int n = ntile * 16 + (lane & 15);
  out[t] = f32_bf16(W[(size_t)k * DIM + n]);
}

// ---------------- degree / counts ----------------
__global__ void deg_kernel(const int* __restrict__ dst, float* __restrict__ deg) {
  int e = blockIdx.x * 256 + threadIdx.x;
  if (e < NE) atomicAdd(&deg[dst[e]], 1.0f);
}
__global__ void invdeg_kernel(float* __restrict__ deg) {
  int n = blockIdx.x * 256 + threadIdx.x;
  if (n < NN) deg[n] = 1.0f / fmaxf(deg[n], 1.0f);
}
__global__ void cnt_kernel(const int* __restrict__ batch, float* __restrict__ cnt) {
  int n = blockIdx.x * 256 + threadIdx.x;
  if (n < NN) atomicAdd(&cnt[batch[n]], 1.0f);
}

// ---------------- edge scatter: agg[dst] += h[src] (L2-resident atomics) ----------------
__global__ __launch_bounds__(256) void scatter_kernel(
    const float* __restrict__ hin, int sin,
    const int* __restrict__ src, const int* __restrict__ dst,
    float* __restrict__ agg) {
  int wave = threadIdx.x >> 5, lane = threadIdx.x & 31;
  int e = blockIdx.x * 8 + wave;
  if (e >= NE) return;
  int s = src[e], d = dst[e];
  const float4 v = *(const float4*)(hin + (size_t)s * sin + lane * 4);
  float* p = agg + (size_t)d * DIM + lane * 4;
  atomicAdd(p + 0, v.x);
  atomicAdd(p + 1, v.y);
  atomicAdd(p + 2, v.z);
  atomicAdd(p + 3, v.w);
}

// A fragment (16-bit A 16x32): lane<16 K{0..7,16..23}, lane>=16 K{8..15,24..31}
// from a bf16 LDS row: two 16B chunks at +0 and +32B from (kk*32 + oa) elems.
__device__ __forceinline__ v8ui load_a_frag_lds(const unsigned short* __restrict__ p) {
  v4ui lo = *(const v4ui*)p;
  v4ui hi = *(const v4ui*)(p + 16);
  return __builtin_shufflevector(lo, hi, 0, 1, 2, 3, 4, 5, 6, 7);
}

// ---------------- SAGE layer: hout = (agg*invdeg)@Wl + bl + hin@Wr ----------------
// 256 thr = 8 waves; block = 16 rows; wave w = column tile w; K=128 -> 4 k-steps.
// A-panels converted f32->bf16 ONCE per block into LDS; hot loop is ds_load + global B load + wmma.
__global__ __launch_bounds__(256) void sage_layer_kernel(
    const float* __restrict__ agg, const float* __restrict__ inv_deg,
    const float* __restrict__ hin, int sin,
    const unsigned short* __restrict__ wl, const unsigned short* __restrict__ wr,
    const float* __restrict__ bias, float* __restrict__ hout, int sout) {
  __shared__ __align__(16) unsigned short sAgg[16 * LDSB];
  __shared__ __align__(16) unsigned short sH[16 * LDSB];
  const int tid = threadIdx.x;
  const int rbase = blockIdx.x * 16;
  {
    // 256 chunks of 8 elems per panel; one chunk per thread per panel
    const int row  = tid >> 4;
    const int col8 = (tid & 15) << 3;
    const float sc = inv_deg[rbase + row];
    *(v4ui*)&sAgg[row * LDSB + col8] =
        cvt8_bf16(agg + (size_t)(rbase + row) * DIM + col8, sc);
    *(v4ui*)&sH[row * LDSB + col8] =
        cvt8_bf16(hin + (size_t)(rbase + row) * sin + col8, 1.0f);
  }
  __syncthreads();

  const int lane  = tid & 31;
  const int ntile = tid >> 5;
  const int r     = lane & 15;
  const int oa    = (lane >> 4) << 3;  // 0 or 8
  v8f acc = {0.f, 0.f, 0.f, 0.f, 0.f, 0.f, 0.f, 0.f};
#pragma unroll
  for (int kk = 0; kk < 4; ++kk) {
    v8ui au = load_a_frag_lds(&sAgg[r * LDSB + kk * 32 + oa]);
    v8ui hu = load_a_frag_lds(&sH[r * LDSB + kk * 32 + oa]);
    v8ui b0 = *(const v8ui*)(wl + (((size_t)ntile * 4 + kk) << 9) + lane * 16);
    v8ui b1 = *(const v8ui*)(wr + (((size_t)ntile * 4 + kk) << 9) + lane * 16);
    acc = __builtin_amdgcn_wmma_f32_16x16x32_bf16(
        false, __builtin_bit_cast(v16bf, au), false, __builtin_bit_cast(v16bf, b0),
        (short)0, acc, false, false);
    acc = __builtin_amdgcn_wmma_f32_16x16x32_bf16(
        false, __builtin_bit_cast(v16bf, hu), false, __builtin_bit_cast(v16bf, b1),
        (short)0, acc, false, false);
  }
  const int col  = ntile * 16 + (lane & 15);
  const int rofs = (lane >> 4) << 3;
  const float b  = bias[col];
#pragma unroll
  for (int rr = 0; rr < 8; ++rr)
    hout[(size_t)(rbase + rofs + rr) * sout + col] = acc[rr] + b;
}

// ---------------- pooling ----------------
__global__ __launch_bounds__(256) void pool_sum_kernel(
    const float* __restrict__ hcat, const int* __restrict__ batch,
    float* __restrict__ pooled) {
  int wave = threadIdx.x >> 5, lane = threadIdx.x & 31;
  int node = blockIdx.x * 8 + wave;
  if (node >= NN) return;
  int g = batch[node];
  const float* hp = hcat + (size_t)node * HC;
  float* pp = pooled + (size_t)g * HC;
#pragma unroll
  for (int j = 0; j < 3; ++j) {
    int c = (j * 32 + lane) * 4;
    float4 v = *(const float4*)(hp + c);
    atomicAdd(pp + c + 0, v.x);
    atomicAdd(pp + c + 1, v.y);
    atomicAdd(pp + c + 2, v.z);
    atomicAdd(pp + c + 3, v.w);
  }
}
__global__ void pool_div_kernel(float* __restrict__ pooled, const float* __restrict__ cnt) {
  int i = blockIdx.x * 256 + threadIdx.x;
  if (i < NG * HC) pooled[i] *= (1.0f / fmaxf(cnt[i / HC], 1.0f));
}

// ---------------- readout layer 1 via WMMA: h1 = relu(pooled @ W1 + b1) ----------------
__global__ __launch_bounds__(256) void mlp1_kernel(
    const float* __restrict__ pooled, const unsigned short* __restrict__ w1p,
    const float* __restrict__ b1, float* __restrict__ h1) {
  const int tid = threadIdx.x;
  const int lane = tid & 31, ntile = tid >> 5;
  const int rbase = blockIdx.x * 16;
  const int r = lane & 15;
  const int oa = (lane >> 4) << 3;
  v8f acc = {0.f, 0.f, 0.f, 0.f, 0.f, 0.f, 0.f, 0.f};
#pragma unroll
  for (int kk = 0; kk < 12; ++kk) {
    const float* pa = pooled + (size_t)(rbase + r) * HC + kk * 32 + oa;
    v8ui au;
#pragma unroll
    for (int j = 0; j < 4; ++j) {
      au[j]     = cvt2_bf16(pa[2 * j], pa[2 * j + 1]);
      au[j + 4] = cvt2_bf16(pa[16 + 2 * j], pa[16 + 2 * j + 1]);
    }
    v8ui bu = *(const v8ui*)(w1p + (((size_t)ntile * 12 + kk) << 9) + lane * 16);
    acc = __builtin_amdgcn_wmma_f32_16x16x32_bf16(
        false, __builtin_bit_cast(v16bf, au), false, __builtin_bit_cast(v16bf, bu),
        (short)0, acc, false, false);
  }
  const int col  = ntile * 16 + r;
  const int rofs = (lane >> 4) << 3;
  const float b  = b1[col];
#pragma unroll
  for (int rr = 0; rr < 8; ++rr)
    h1[(size_t)(rbase + rofs + rr) * DIM + col] = fmaxf(acc[rr] + b, 0.0f);
}

// ---------------- readout layer 2 (tiny): out = h1 @ W2 + b2 ----------------
__global__ void mlp2_kernel(const float* __restrict__ h1, const float* __restrict__ W2,
                            const float* __restrict__ b2, float* __restrict__ out) {
  int id = blockIdx.x * 256 + threadIdx.x;
  if (id >= NG * NT) return;
  int g = id / NT, t = id % NT;
  const float* hp = h1 + (size_t)g * DIM;
  float s = b2[t];
#pragma unroll 8
  for (int d = 0; d < DIM; ++d) s = fmaf(hp[d], W2[d * NT + t], s);
  out[id] = s;
}

// ---------------- host side ----------------
extern "C" void kernel_launch(void* const* d_in, const int* in_sizes, int n_in,
                              void* d_out, int out_size, void* d_ws, size_t ws_size,
                              hipStream_t stream) {
  (void)in_sizes; (void)n_in; (void)out_size; (void)ws_size;
  const float* x     = (const float*)d_in[0];
  const int*   eidx  = (const int*)d_in[1];
  const int*   src   = eidx;
  const int*   dst   = eidx + NE;
  const int*   batch = (const int*)d_in[2];
  const float* Wl    = (const float*)d_in[3];
  const float* bl    = (const float*)d_in[4];
  const float* Wr    = (const float*)d_in[5];
  const float* W1    = (const float*)d_in[6];
  const float* b1    = (const float*)d_in[7];
  const float* W2    = (const float*)d_in[8];
  const float* b2    = (const float*)d_in[9];
  float* out = (float*)d_out;

  // 256B-aligned bump allocation inside d_ws (~99 MB total)
  char* wsb = (char*)d_ws;
  size_t off = 0;
  auto alloc = [&](size_t bytes) -> char* {
    char* p = wsb + off;
    off = (off + bytes + 255) & ~(size_t)255;
    return p;
  };
  float* inv_deg = (float*)alloc((size_t)NN * 4);
  float* cnt     = (float*)alloc((size_t)NG * 4);
  float* agg     = (float*)alloc((size_t)NN * DIM * 4);
  float* hcat    = (float*)alloc((size_t)NN * HC * 4);
  float* pooled  = (float*)alloc((size_t)NG * HC * 4);
  float* h1      = (float*)alloc((size_t)NG * DIM * 4);
  unsigned short* wlp = (unsigned short*)alloc((size_t)NL * DIM * DIM * 2);
  unsigned short* wrp = (unsigned short*)alloc((size_t)NL * DIM * DIM * 2);
  unsigned short* w1p = (unsigned short*)alloc((size_t)HC * DIM * 2);

  hipMemsetAsync(inv_deg, 0, (size_t)NN * 4, stream);
  hipMemsetAsync(cnt, 0, (size_t)NG * 4, stream);
  hipMemsetAsync(pooled, 0, (size_t)NG * HC * 4, stream);

  // pre-pack weights to bf16 B-fragments (L2-resident, reused by every block)
  for (int i = 0; i < NL; ++i) {
    pack_b_kernel<<<64, 256, 0, stream>>>(Wl + (size_t)i * DIM * DIM, DIM, wlp + (size_t)i * DIM * DIM);
    pack_b_kernel<<<64, 256, 0, stream>>>(Wr + (size_t)i * DIM * DIM, DIM, wrp + (size_t)i * DIM * DIM);
  }
  pack_b_kernel<<<192, 256, 0, stream>>>(W1, HC, w1p);

  deg_kernel<<<(NE + 255) / 256, 256, 0, stream>>>(dst, inv_deg);
  invdeg_kernel<<<(NN + 255) / 256, 256, 0, stream>>>(inv_deg);
  cnt_kernel<<<(NN + 255) / 256, 256, 0, stream>>>(batch, cnt);

  const float* hin = x;
  int sin = DIM;
  for (int i = 0; i < NL; ++i) {
    hipMemsetAsync(agg, 0, (size_t)NN * DIM * 4, stream);
    scatter_kernel<<<NE / 8, 256, 0, stream>>>(hin, sin, src, dst, agg);
    sage_layer_kernel<<<NN / 16, 256, 0, stream>>>(
        agg, inv_deg, hin, sin,
        wlp + (size_t)i * DIM * DIM, wrp + (size_t)i * DIM * DIM,
        bl + (size_t)i * DIM, hcat + (size_t)i * DIM, HC);
    hin = hcat + (size_t)i * DIM;
    sin = HC;
  }

  pool_sum_kernel<<<NN / 8, 256, 0, stream>>>(hcat, batch, pooled);
  pool_div_kernel<<<(NG * HC + 255) / 256, 256, 0, stream>>>(pooled, cnt);
  mlp1_kernel<<<NG / 16, 256, 0, stream>>>(pooled, w1p, b1, h1);
  mlp2_kernel<<<(NG * NT + 255) / 256, 256, 0, stream>>>(h1, W2, b2, out);
}
